// ResidualConvLSTM2D_71322226917430
// MI455X (gfx1250) — matrix operand order repaired and checked by reading, not compiled, not run on hardware
//
#include <hip/hip_runtime.h>
#include <hip/hip_bf16.h>
#include <math.h>

// ---- dims ----
#define BB   4
#define TT   16
#define HH   64
#define WW   64
#define CIN  32
#define FF   64    // hidden channels
#define G4   256   // 4*FF gate channels

typedef __attribute__((ext_vector_type(16))) __bf16 v16bf;
typedef __attribute__((ext_vector_type(8)))  float  v8f;

union BV { v16bf v; uint4 q[2]; };

__device__ inline v8f wmma_bf16(v16bf a, v16bf b, v8f c) {
    // D = A(16x32 bf16) * B(32x16 bf16) + C(16x16 f32)
    return __builtin_amdgcn_wmma_f32_16x16x32_bf16(
        /*neg_a=*/false, a, /*neg_b=*/false, b,
        /*c_mod=*/(short)0, c, /*reuse_a=*/false, /*reuse_b=*/false);
}

__device__ inline float hsig(float z) {
    return fminf(fmaxf(0.2f * z + 0.5f, 0.0f), 1.0f);
}

// ---------------- prep kernels ----------------

__global__ void k_cvt_x(const float* __restrict__ x, __bf16* __restrict__ xb, int n) {
    int i = blockIdx.x * blockDim.x + threadIdx.x;
    if (i < n) xb[i] = (__bf16)x[i];
}

// Pack weights into per-lane B-operand layout:
//   packed[((chunk*2 + half)*N + n)*16 + j] = W[K = (j&7) + half*8 + (j>>3)*16 within chunk][n]
__global__ void k_pack(const float* __restrict__ Wx, const float* __restrict__ Wh,
                       const float* __restrict__ Wp,
                       __bf16* __restrict__ Wxp, __bf16* __restrict__ Whp,
                       __bf16* __restrict__ Wpp) {
    int o = blockIdx.x * blockDim.x + threadIdx.x;
    if (o < 73728) {                           // Wx: 9 chunks (taps), K=cin(32), N=256
        int j = o & 15, n = (o >> 4) & 255, h = (o >> 12) & 1, c = o >> 13; // c=tap 0..8
        Wxp[o] = (__bf16)Wx[(c * 32 + h * 8 + (j & 7) + ((j >> 3) << 4)) * 256 + n];
    } else if (o < 73728 + 147456) {           // Wh: 18 chunks (tap*2+blk), N=256
        int p = o - 73728;
        int j = p & 15, n = (p >> 4) & 255, h = (p >> 12) & 1, c = p >> 13; // c 0..17
        int tap = c >> 1, blk = c & 1;
        Whp[p] = (__bf16)Wh[(tap * 64 + blk * 32 + h * 8 + (j & 7) + ((j >> 3) << 4)) * 256 + n];
    } else if (o < 73728 + 147456 + 2048) {    // Wp: 1 chunk, K=32, N=64
        int p = o - 73728 - 147456;
        int j = p & 15, n = (p >> 4) & 63, h = p >> 10;
        Wpp[p] = (__bf16)Wp[(h * 8 + (j & 7) + ((j >> 3) << 4)) * 64 + n];
    }
}

__global__ void k_init(__bf16* __restrict__ hb0, float* __restrict__ cbuf, int n) {
    int i = blockIdx.x * blockDim.x + threadIdx.x;
    if (i < n) { hb0[i] = (__bf16)0.0f; cbuf[i] = 0.0f; }
}

// ---------------- residual: out = x (1x1 conv) Wp + bp, all B*T ----------------

__global__ __launch_bounds__(256) void k_residual(const __bf16* __restrict__ xb,
                                                  const __bf16* __restrict__ Wpp,
                                                  const float* __restrict__ bp,
                                                  float* __restrict__ out) {
    int lane = threadIdx.x & 31, wave = threadIdx.x >> 5;
    int gw = blockIdx.x * 8 + wave;            // 65536 waves = 16384 mtiles * 4 ftiles
    int mt = gw >> 2, ftile = gw & 3;
    int bt  = mt >> 8;                          // b*T + t  (0..63)
    int rem = mt & 255;
    int y = rem >> 2, x0 = (rem & 3) << 4;
    int m = lane & 15, half = lane >> 4, nl = lane & 15;

    // A: 16 positions x K=32 (cin), per-lane two contiguous 16B loads
    const uint4* ap = (const uint4*)(xb + ((size_t)bt * HH * WW + (size_t)y * WW + (x0 + m)) * CIN + half * 8);
    BV a; a.q[0] = ap[0]; a.q[1] = ap[2];

    const uint4* bq = (const uint4*)(Wpp + ((size_t)half * 64 + ftile * 16 + nl) * 16);
    BV bm; bm.q[0] = bq[0]; bm.q[1] = bq[1];

    v8f acc = {};
    acc = wmma_bf16(a.v, bm.v, acc);

    int f = ftile * 16 + nl;
    float bias = bp[f];
#pragma unroll
    for (int r = 0; r < 8; ++r) {
        int pm = r + half * 8;
        size_t oidx = ((size_t)bt * HH * WW + (size_t)y * WW + (x0 + pm)) * FF + f;
        out[oidx] = acc[r] + bias;
    }
}

// ---------------- fused ConvLSTM step ----------------
// z = conv(x_t, Wx) + conv(h_prev, Wh) + b  (via 27 K-chunks x 4 gate tiles of WMMA)
// then gates, c/h update, h -> bf16 ping-pong buffer, out += h.

__global__ __launch_bounds__(256) void k_step(const __bf16* __restrict__ xb,
                                              const __bf16* __restrict__ Wxp,
                                              const __bf16* __restrict__ Whp,
                                              const float* __restrict__ bias,
                                              const __bf16* __restrict__ hprev,
                                              __bf16* __restrict__ hnext,
                                              float* __restrict__ cbuf,
                                              float* __restrict__ out,
                                              int t) {
    int lane = threadIdx.x & 31, wave = threadIdx.x >> 5;
    int gw = blockIdx.x * 8 + wave;            // 4096 waves = 1024 mtiles * 4 ftiles
    int mt = gw >> 2, ftile = gw & 3;
    int b   = mt >> 8;
    int rem = mt & 255;
    int y = rem >> 2, x0 = (rem & 3) << 4;
    int m = lane & 15, half = lane >> 4, nl = lane & 15;
    int xw = x0 + m;

    v8f acc[4] = { {}, {}, {}, {} };
    uint4 z4 = {0u, 0u, 0u, 0u};

    // ---- 9 chunks from x_t (tap, K = cin 0..31) ----
    const __bf16* xbase = xb + (size_t)(b * TT + t) * HH * WW * CIN;
    for (int tap = 0; tap < 9; ++tap) {
        int yy = y + tap / 3 - 1;
        int xx = xw + tap % 3 - 1;
        BV a;
        if ((unsigned)yy < HH && (unsigned)xx < WW) {
            const uint4* ap = (const uint4*)(xbase + ((size_t)yy * WW + xx) * CIN + half * 8);
            a.q[0] = ap[0]; a.q[1] = ap[2];
        } else {
            a.q[0] = z4; a.q[1] = z4;
        }
        const __bf16* wb = Wxp + ((size_t)(tap * 2 + half) * 256) * 16;
#pragma unroll
        for (int g = 0; g < 4; ++g) {
            const uint4* bq = (const uint4*)(wb + (size_t)(g * 64 + ftile * 16 + nl) * 16);
            BV bm; bm.q[0] = bq[0]; bm.q[1] = bq[1];
            acc[g] = wmma_bf16(a.v, bm.v, acc[g]);
        }
    }

    // ---- 18 chunks from h_prev (tap x 2 cin-blocks of 32) ----
    const __bf16* hbase = hprev + (size_t)b * HH * WW * FF;
    for (int tap = 0; tap < 9; ++tap) {
        int yy = y + tap / 3 - 1;
        int xx = xw + tap % 3 - 1;
        bool valid = ((unsigned)yy < HH) && ((unsigned)xx < WW);
        const __bf16* hp = hbase + ((size_t)(yy * WW + xx)) * FF;
#pragma unroll
        for (int blk = 0; blk < 2; ++blk) {
            BV a;
            if (valid) {
                const uint4* ap = (const uint4*)(hp + blk * 32 + half * 8);
                a.q[0] = ap[0]; a.q[1] = ap[2];
            } else {
                a.q[0] = z4; a.q[1] = z4;
            }
            int chunk = tap * 2 + blk;
            const __bf16* wb = Whp + ((size_t)(chunk * 2 + half) * 256) * 16;
#pragma unroll
            for (int g = 0; g < 4; ++g) {
                const uint4* bq = (const uint4*)(wb + (size_t)(g * 64 + ftile * 16 + nl) * 16);
                BV bm; bm.q[0] = bq[0]; bm.q[1] = bq[1];
                acc[g] = wmma_bf16(a.v, bm.v, acc[g]);
            }
        }
    }

    // ---- gate nonlinearity + state update ----
    int f = ftile * 16 + nl;
    float bi = bias[f], bf = bias[64 + f], bc = bias[128 + f], bo = bias[192 + f];
#pragma unroll
    for (int r = 0; r < 8; ++r) {
        int pm = r + half * 8;
        int xwp = x0 + pm;
        float zi = acc[0][r] + bi;
        float zf = acc[1][r] + bf;
        float zc = acc[2][r] + bc;
        float zo = acc[3][r] + bo;
        float ig = hsig(zi), fg = hsig(zf), og = hsig(zo);
        size_t sidx = ((size_t)(b * HH + y) * WW + xwp) * FF + f;
        float cn = fg * cbuf[sidx] + ig * tanhf(zc);
        cbuf[sidx] = cn;
        float hn = og * tanhf(cn);
        hnext[sidx] = (__bf16)hn;
        size_t oidx = (((size_t)(b * TT + t) * HH + y) * WW + xwp) * FF + f;
        out[oidx] += hn;   // residual already stored by k_residual
    }
}

// ---------------- host launch ----------------

extern "C" void kernel_launch(void* const* d_in, const int* in_sizes, int n_in,
                              void* d_out, int out_size, void* d_ws, size_t ws_size,
                              hipStream_t stream) {
    (void)in_sizes; (void)n_in; (void)out_size; (void)ws_size;
    const float* x  = (const float*)d_in[0];
    const float* Wx = (const float*)d_in[1];
    const float* Wh = (const float*)d_in[2];
    const float* b  = (const float*)d_in[3];
    const float* Wp = (const float*)d_in[4];
    const float* bp = (const float*)d_in[5];
    float* out = (float*)d_out;

    char* ws = (char*)d_ws;
    __bf16* xb   = (__bf16*)(ws);                 // 8,388,608 bf16 = 16 MiB (64 MiB reserved)
    __bf16* Wxp  = (__bf16*)(ws + 67108864);      // 73,728 bf16
    __bf16* Whp  = (__bf16*)(ws + 67256320);      // 147,456 bf16
    __bf16* Wpp  = (__bf16*)(ws + 67551232);      // 2,048 bf16
    __bf16* hb0  = (__bf16*)(ws + 67555328);      // 1,048,576 bf16
    __bf16* hb1  = (__bf16*)(ws + 69652480);      // 1,048,576 bf16
    float*  cbuf = (float*) (ws + 71749632);      // 1,048,576 f32

    const int nx = BB * TT * HH * WW * CIN;       // 8,388,608
    k_cvt_x<<<nx / 256, 256, 0, stream>>>(x, xb, nx);
    k_pack<<<(73728 + 147456 + 2048 + 255) / 256, 256, 0, stream>>>(Wx, Wh, Wp, Wxp, Whp, Wpp);
    k_init<<<(BB * HH * WW * FF) / 256, 256, 0, stream>>>(hb0, cbuf, BB * HH * WW * FF);

    // residual branch for all (B,T): out = x . Wp + bp
    k_residual<<<8192, 256, 0, stream>>>(xb, Wpp, bp, out);

    // sequential ConvLSTM steps (ping-pong h buffers)
    for (int t = 0; t < TT; ++t) {
        const __bf16* hr = (t & 1) ? hb1 : hb0;
        __bf16*       hw = (t & 1) ? hb0 : hb1;
        k_step<<<512, 256, 0, stream>>>(xb, Wxp, Whp, b, hr, hw, cbuf, out, t);
    }
}
